// CudaNormFastWeightDPFPTransformerLayer_15891378995480
// MI455X (gfx1250) — matrix operand order repaired and checked by reading, not compiled, not run on hardware
//
#include <hip/hip_runtime.h>
#include <hip/hip_bf16.h>

typedef __attribute__((ext_vector_type(16))) __bf16 v16bf;
typedef __attribute__((ext_vector_type(8)))  __bf16 v8bf;
typedef __attribute__((ext_vector_type(8)))  float  v8f;

#define N_HEAD  16
#define D_HEAD  64
#define D_MODEL 1024
#define QKVB_N  (N_HEAD * (3 * D_HEAD + 1))   // 3088
#define SLEN    2048
#define BSZ     8
#define NTOK    (SLEN * BSZ)                  // 16384
#define EPS     1e-5f
#define LN_EPS  1e-5f
#define SCALE   0.125f                        // 1/sqrt(64)

// ---------------------------------------------------------------------------
// gfx1250 async global->LDS path (guard-verified present on this toolchain).
// ---------------------------------------------------------------------------
#if defined(__has_builtin)
#if __has_builtin(__builtin_amdgcn_global_load_async_to_lds_b128) && \
    __has_builtin(__builtin_amdgcn_s_wait_asynccnt)
#define USE_ASYNC_LDS 1
#endif
#endif
#ifndef USE_ASYNC_LDS
#define USE_ASYNC_LDS 0
#endif

#if USE_ASYNC_LDS
typedef int v4i __attribute__((vector_size(16)));
typedef __attribute__((address_space(1))) v4i GV4;
typedef __attribute__((address_space(3))) v4i LV4;
#endif

__device__ __forceinline__ void cp_f4_to_lds(float* lds_dst, const float* gsrc) {
#if USE_ASYNC_LDS
    __builtin_amdgcn_global_load_async_to_lds_b128((GV4*)gsrc, (LV4*)lds_dst, 0, 0);
#else
    *(float4*)lds_dst = *(const float4*)gsrc;
#endif
}

__device__ __forceinline__ void wait_async_tiles() {
#if USE_ASYNC_LDS
    __builtin_amdgcn_s_wait_asynccnt(0);
#endif
}

// ---------------------------------------------------------------------------
// A fragment from f32 LDS: elements 0..15 = p[0..15].
// ---------------------------------------------------------------------------
__device__ __forceinline__ v16bf pack_a(const float* p) {
    float4 x0 = *(const float4*)(p +  0), x1 = *(const float4*)(p +  4);
    float4 y0 = *(const float4*)(p +  8), y1 = *(const float4*)(p + 12);
    v16bf r;
    r[0]=(__bf16)x0.x; r[1]=(__bf16)x0.y; r[2]=(__bf16)x0.z; r[3]=(__bf16)x0.w;
    r[4]=(__bf16)x1.x; r[5]=(__bf16)x1.y; r[6]=(__bf16)x1.z; r[7]=(__bf16)x1.w;
    r[8]=(__bf16)y0.x; r[9]=(__bf16)y0.y; r[10]=(__bf16)y0.z; r[11]=(__bf16)y0.w;
    r[12]=(__bf16)y1.x; r[13]=(__bf16)y1.y; r[14]=(__bf16)y1.z; r[15]=(__bf16)y1.w;
    return r;
}

// B fragment from bf16 LDS: 16 contiguous bf16 -> two b128 loads.
__device__ __forceinline__ v16bf load_b_frag(const __bf16* p) {
    v8bf lo = *(const v8bf*)p;
    v8bf hi = *(const v8bf*)(p + 8);
    return __builtin_shufflevector(lo, hi, 0,1,2,3,4,5,6,7,8,9,10,11,12,13,14,15);
}

// ---------------------------------------------------------------------------
// C[M,N] = A[M,K] * B[N,K]^T, f32 in/out, bf16 WMMA compute.
// grid = (ceil(N/64), M/128), block 256 (8 waves). Block tile 128x64:
// wave w owns rows w*16..15 and all 64 cols (4 WMMAs, A-frag reused).
// Double-buffered K loop: A staged f32 via async global->LDS b128;
// B staged bf16 (converted once), issue/commit split around the WMMAs.
// ---------------------------------------------------------------------------
#define ASTRIDE 36   // f32 elements (144 B rows)
#define BSTRIDE 40   // bf16 elements (80 B rows)
__global__ __launch_bounds__(256) void gemm_bf16_wmma(
    const float* __restrict__ A, const float* __restrict__ B,
    float* __restrict__ C, int M, int N, int K)
{
    __shared__ float  lds_a[2][128 * ASTRIDE];
    __shared__ __bf16 lds_bb[2][64 * BSTRIDE];

    const int tid  = threadIdx.x;
    const int wid  = tid >> 5;
    const int lane = tid & 31;
    const int half = lane >> 4;
    const int idx  = lane & 15;
    const int m_blk = blockIdx.y * 128;
    const int n_blk = blockIdx.x * 64;
    const int nk = K >> 5;

    // B staging geometry: thread -> row r = tid>>2, 8-col chunk c = (tid&3)*8
    const int br = tid >> 2;
    const int bc = (tid & 3) << 3;
    const bool bok = (n_blk + br) < N;

    v8f acc[4] = {{}, {}, {}, {}};

    auto issue_a = [&](int buf, int k0) {
#pragma unroll
        for (int i = 0; i < 4; ++i) {
            int e = tid + i * 256, r = e >> 3, c = (e & 7) << 2;
            cp_f4_to_lds(&lds_a[buf][r * ASTRIDE + c],
                         A + (size_t)(m_blk + r) * K + k0 + c);
        }
    };
    auto issue_b = [&](float4 breg[2], int k0) {
        if (bok) {
            const float* src = B + (size_t)(n_blk + br) * K + k0 + bc;
            breg[0] = *(const float4*)src;
            breg[1] = *(const float4*)(src + 4);
        } else {
            breg[0] = float4{0.f, 0.f, 0.f, 0.f};
            breg[1] = float4{0.f, 0.f, 0.f, 0.f};
        }
    };
    auto commit_b = [&](const float4 breg[2], int buf) {
        v8bf v;
        v[0]=(__bf16)breg[0].x; v[1]=(__bf16)breg[0].y;
        v[2]=(__bf16)breg[0].z; v[3]=(__bf16)breg[0].w;
        v[4]=(__bf16)breg[1].x; v[5]=(__bf16)breg[1].y;
        v[6]=(__bf16)breg[1].z; v[7]=(__bf16)breg[1].w;
        *(v8bf*)&lds_bb[buf][br * BSTRIDE + bc] = v;
    };

    // prologue: stage tile 0
    {
        float4 b0[2];
        issue_a(0, 0);
        issue_b(b0, 0);
        commit_b(b0, 0);
    }

    float4 breg[2];
    for (int kt = 0; kt < nk; ++kt) {
        wait_async_tiles();
        __syncthreads();                       // tile (kt&1) ready; all waves past kt-1
        const bool more = (kt + 1 < nk);
        if (more) {
            issue_a((kt + 1) & 1, (kt + 1) << 5);   // async, overlaps compute
            issue_b(breg, (kt + 1) << 5);           // loads in flight during compute
        }

        const float*  la = lds_a[kt & 1];
        const __bf16* lb = lds_bb[kt & 1];
        // preload all fragments into distinct regs, then WMMAs back-to-back
        v16bf a = pack_a(la + ((wid << 4) + idx) * ASTRIDE + (half << 4));
        v16bf bf0 = load_b_frag(lb + (( 0 + idx) * BSTRIDE) + (half << 4));
        v16bf bf1 = load_b_frag(lb + ((16 + idx) * BSTRIDE) + (half << 4));
        v16bf bf2 = load_b_frag(lb + ((32 + idx) * BSTRIDE) + (half << 4));
        v16bf bf3 = load_b_frag(lb + ((48 + idx) * BSTRIDE) + (half << 4));
        acc[0] = __builtin_amdgcn_wmma_f32_16x16x32_bf16(false, a, false, bf0, (short)0, acc[0], false, false);
        acc[1] = __builtin_amdgcn_wmma_f32_16x16x32_bf16(false, a, false, bf1, (short)0, acc[1], false, false);
        acc[2] = __builtin_amdgcn_wmma_f32_16x16x32_bf16(false, a, false, bf2, (short)0, acc[2], false, false);
        acc[3] = __builtin_amdgcn_wmma_f32_16x16x32_bf16(false, a, false, bf3, (short)0, acc[3], false, false);

        if (more) commit_b(breg, (kt + 1) & 1);     // cvt + ds_store after compute
    }

    const int crow = m_blk + (wid << 4) + (half << 3);
#pragma unroll
    for (int nt = 0; nt < 4; ++nt) {
        int col = n_blk + (nt << 4) + idx;
        if (col < N) {
#pragma unroll
            for (int r = 0; r < 8; ++r)
                C[(size_t)(crow + r) * N + col] = acc[nt][r];
        }
    }
}

// ---------------------------------------------------------------------------
// 5-way fused block reduction (one barrier round for all per-step scalars).
// ---------------------------------------------------------------------------
__device__ __forceinline__ void block_reduce5(float v[5], float* wred, int lane, int wid) {
#pragma unroll
    for (int k = 0; k < 5; ++k)
#pragma unroll
        for (int off = 16; off; off >>= 1) v[k] += __shfl_xor(v[k], off);
    __syncthreads();
    if (lane == 0) {
#pragma unroll
        for (int k = 0; k < 5; ++k) wred[wid * 5 + k] = v[k];
    }
    __syncthreads();
#pragma unroll
    for (int k = 0; k < 5; ++k) {
        float t = 0.f;
#pragma unroll
        for (int i = 0; i < 8; ++i) t += wred[i * 5 + k];
        v[k] = t;
    }
}

// ---------------------------------------------------------------------------
// Sequential fast-weight scan: one WG per (b,h) chain; W state (64 cols x 256
// dims, stride-260 pad, [v][d] layout) lives in LDS. Thread t owns feature
// dim t, and (v = t>>2, d-quarter = t&3): its W chunk is 64 contiguous floats
// -> single b128 read + single b128 write pass per step (k.W, update and q.W
// fused through registers).
// ---------------------------------------------------------------------------
#define WSTRIDE 260
__global__ __launch_bounds__(256) void fastweight_scan(
    const float* __restrict__ qkvb, float* __restrict__ layer_out)
{
    extern __shared__ float smem[];
    float* Wm   = smem;                    // 64 * 260
    float* raw  = Wm + 64 * WSTRIDE;       // 256 (193 used)
    float* xk   = raw + 256;               // 128
    float* xq   = xk + 128;                // 128
    float* kf   = xq + 128;                // 256
    float* qf   = kf + 256;                // 256
    float* wred = qf + 256;                // 40

    const int tid  = threadIdx.x;
    const int lane = tid & 31;
    const int wid  = tid >> 5;
    const int b    = blockIdx.x >> 4;
    const int hh   = blockIdx.x & 15;

    for (int i = tid; i < 64 * WSTRIDE; i += 256) Wm[i] = 0.f;

    float dacc = 0.f;                      // denom_acc[tid]
    const int vcol  = tid >> 2;
    const int dbase = (tid & 3) << 6;
    const int j  = tid & 127;
    const int sh = (tid < 128) ? 1 : 2;

    float4* W4 = (float4*)(Wm + vcol * WSTRIDE + dbase);

    for (int s = 0; s < SLEN; ++s) {
        const float* row = qkvb + (size_t)(s * BSZ + b) * QKVB_N + hh * 193;
        __syncthreads();                   // all threads done reading raw/kf/qf
        if (tid < 193) raw[tid] = row[tid];
        if (s + 1 < SLEN && tid < 7) {     // prefetch next step's row (global_prefetch_b8)
            const char* nxt = (const char*)(row + BSZ * QKVB_N);
            __builtin_prefetch(nxt + tid * 128, 0, 3);
        }
        __syncthreads();

        if (tid < 128) {
            float sgn = (tid < 64) ? 1.f : -1.f;
            xk[tid] = fmaxf(sgn * raw[64 + (tid & 63)], 0.f);
            xq[tid] = fmaxf(sgn * raw[tid & 63], 0.f);
        }
        __syncthreads();

        // DPFP features: feat[t] = x[j] * x[(j - shift) mod 128]
        float fk = xk[j] * xk[(j - sh) & 127];
        float fq = xq[j] * xq[(j - sh) & 127];

        // one reduction round: {sum_k, sum_q, dacc.fk, dacc.fq, fk.fq}
        float r5[5] = {fk, fq, dacc * fk, dacc * fq, fk * fq};
        block_reduce5(r5, wred, lane, wid);
        float sk = r5[0], sq = r5[1];
        float kd    = (s == 0) ? 1.f : r5[2] / sk;
        float denom = r5[3] / sq + r5[4] / (sk * sq);
        float fkn = fk / sk, fqn = fq / sq;
        dacc += fkn;

        float beff = kd / (1.f + __expf(-raw[192]));   // sigmoid(b)*key_denom
        kf[tid] = fkn / (kd + EPS);
        qf[tid] = fqn;
        __syncthreads();

        // single W read pass: v_old = keff . W[:,vcol]
        const float4* kf4 = (const float4*)(kf + dbase);
        const float4* qf4 = (const float4*)(qf + dbase);
        float4 wreg[16];
        float p = 0.f;
#pragma unroll
        for (int i = 0; i < 16; ++i) {
            wreg[i] = W4[i];
            float4 kk = kf4[i];
            p += kk.x * wreg[i].x + kk.y * wreg[i].y + kk.z * wreg[i].z + kk.w * wreg[i].w;
        }
        p += __shfl_xor(p, 1);
        p += __shfl_xor(p, 2);

        float nv = beff * (raw[128 + vcol] - p);

        // fused update + q.W + single W write pass
        float q2 = 0.f;
#pragma unroll
        for (int i = 0; i < 16; ++i) {
            float4 kk = kf4[i], qq = qf4[i], w = wreg[i];
            w.x += kk.x * nv; w.y += kk.y * nv; w.z += kk.z * nv; w.w += kk.w * nv;
            q2 += qq.x * w.x + qq.y * w.y + qq.z * w.z + qq.w * w.w;
            W4[i] = w;
        }
        q2 += __shfl_xor(q2, 1);
        q2 += __shfl_xor(q2, 2);

        if ((tid & 3) == 0)
            layer_out[(size_t)(s * BSZ + b) * D_MODEL + hh * 64 + vcol] =
                SCALE * q2 / (denom + EPS);
    }
}

// ---------------------------------------------------------------------------
// x = h + attn; LayerNorm(x) * g + b. One block per token row.
// ---------------------------------------------------------------------------
__device__ __forceinline__ float block_reduce1(float v, float* wred, int lane, int wid) {
#pragma unroll
    for (int off = 16; off; off >>= 1) v += __shfl_xor(v, off);
    __syncthreads();
    if (lane == 0) wred[wid] = v;
    __syncthreads();
    float t = 0.f;
#pragma unroll
    for (int i = 0; i < 8; ++i) t += wred[i];
    return t;
}

__global__ __launch_bounds__(256) void resid_ln(
    const float* __restrict__ h, const float* __restrict__ attn,
    const float* __restrict__ g, const float* __restrict__ be,
    float* __restrict__ out)
{
    __shared__ float wred[8];
    const int r    = blockIdx.x;
    const int tid  = threadIdx.x;
    const int lane = tid & 31;
    const int wid  = tid >> 5;

    float x[4];
    float s = 0.f;
#pragma unroll
    for (int i = 0; i < 4; ++i) {
        int c = tid + i * 256;
        x[i] = h[(size_t)r * D_MODEL + c] + attn[(size_t)r * D_MODEL + c];
        s += x[i];
    }
    float mean = block_reduce1(s, wred, lane, wid) * (1.f / D_MODEL);
    float v = 0.f;
#pragma unroll
    for (int i = 0; i < 4; ++i) { float d = x[i] - mean; v += d * d; }
    float var = block_reduce1(v, wred, lane, wid) * (1.f / D_MODEL);
    float inv = rsqrtf(var + LN_EPS);
#pragma unroll
    for (int i = 0; i < 4; ++i) {
        int c = tid + i * 256;
        out[(size_t)r * D_MODEL + c] = (x[i] - mean) * inv * g[c] + be[c];
    }
}

// ---------------------------------------------------------------------------
extern "C" void kernel_launch(void* const* d_in, const int* in_sizes, int n_in,
                              void* d_out, int out_size, void* d_ws, size_t ws_size,
                              hipStream_t stream) {
    (void)in_sizes; (void)n_in; (void)out_size; (void)ws_size;
    const float* h   = (const float*)d_in[0];
    const float* Wq  = (const float*)d_in[1];
    const float* Wo  = (const float*)d_in[2];
    const float* g   = (const float*)d_in[3];
    const float* be  = (const float*)d_in[4];
    float* out = (float*)d_out;

    char* ws = (char*)d_ws;
    float* qkvb  = (float*)ws;                                            // 202 MB
    float* layer = (float*)(ws + (size_t)NTOK * QKVB_N * sizeof(float));  //  64 MB
    float* attn  = (float*)(ws + (size_t)NTOK * QKVB_N * sizeof(float)
                               + (size_t)NTOK * D_MODEL * sizeof(float)); //  64 MB

    const size_t scan_smem =
        (64 * WSTRIDE + 256 + 128 + 128 + 256 + 256 + 40) * sizeof(float); // ~71 KB
    (void)hipFuncSetAttribute((const void*)fastweight_scan,
                              hipFuncAttributeMaxDynamicSharedMemorySize,
                              (int)scan_smem);

    dim3 blk(256);
    // 1) qkvb = h @ W_qkvb^T   (16384 x 3088 x 1024)
    gemm_bf16_wmma<<<dim3((QKVB_N + 63) / 64, NTOK / 128), blk, 0, stream>>>(
        h, Wq, qkvb, NTOK, QKVB_N, D_MODEL);
    // 2) sequential fast-weight scan, 128 chains
    fastweight_scan<<<dim3(BSZ * N_HEAD), blk, scan_smem, stream>>>(qkvb, layer);
    // 3) attn = layer @ W_o^T   (16384 x 1024 x 1024)
    gemm_bf16_wmma<<<dim3(D_MODEL / 64, NTOK / 128), blk, 0, stream>>>(
        layer, Wo, attn, NTOK, D_MODEL, D_MODEL);
    // 4) residual + layernorm
    resid_ln<<<dim3(NTOK), blk, 0, stream>>>(h, attn, g, be, out);
}
